// RCNNLayer_37426345017666
// MI455X (gfx1250) — matrix operand order repaired and checked by reading, not compiled, not run on hardware
//
#include <hip/hip_runtime.h>

#define B_   32
#define S_   512
#define IN_  256
#define H_   512
#define OUT_ 256
#define CTX_ 1280
#define HP_  520   // padded LDS row (bf16): 1040B -> 4-bank row stride, conflict-free

typedef __attribute__((ext_vector_type(16))) __bf16 v16bf;
typedef __attribute__((ext_vector_type(8)))  __bf16 v8bf;
typedef __attribute__((ext_vector_type(8)))  float  v8f;

static __device__ __forceinline__ v16bf cat8(v8bf lo, v8bf hi) {
  return __builtin_shufflevector(lo, hi, 0,1,2,3,4,5,6,7,8,9,10,11,12,13,14,15);
}

// A-fragment (16x32 bf16, MxK): lane holds row = lane%16, K runs
// [ks..ks+7] and [ks+16..ks+23] with ks = 8*(lane/16). p points at run start.
static __device__ __forceinline__ v16bf load_a16(const __bf16* p) {
  v8bf lo = *(const v8bf*)p;
  v8bf hi = *(const v8bf*)(p + 16);
  return cat8(lo, hi);
}

static __device__ __forceinline__ v16bf zero16() {
  v16bf z;
#pragma unroll
  for (int i = 0; i < 16; ++i) z[i] = (__bf16)0.0f;
  return z;
}

// B-fragment from packed weights: layout [nt][kc][lane][16] -> each wave load is
// one contiguous 1KB burst (32 lanes x 32B).
static __device__ __forceinline__ v16bf load_bP(const __bf16* Wp, int nkc, int nt, int kc) {
  const int lane = threadIdx.x & 31;
  return *(const v16bf*)(Wp + ((((size_t)nt * nkc + kc) * 32 + lane) << 4));
}

static __device__ __forceinline__ v8f wmma_bf16(v16bf a, v16bf b, v8f c) {
  return __builtin_amdgcn_wmma_f32_16x16x32_bf16(false, a, false, b, (short)0, c,
                                                 false, false);
}

// Fast sigmoid: v_exp_f32 (base-2) + v_rcp_f32 instead of the IEEE divide sequence.
static __device__ __forceinline__ float sigm(float x) {
#if __has_builtin(__builtin_amdgcn_rcpf) && __has_builtin(__builtin_amdgcn_exp2f)
  return __builtin_amdgcn_rcpf(1.0f + __builtin_amdgcn_exp2f(x * -1.442695041f));
#else
  return 1.0f / (1.0f + __expf(-x));
#endif
}

// gfx1250 has a native v_tanh_f32 TRANS op.
static __device__ __forceinline__ float fast_tanh(float x) {
#if __has_builtin(__builtin_amdgcn_tanhf)
  return __builtin_amdgcn_tanhf(x);
#else
  return 2.0f * sigm(2.0f * x) - 1.0f;
#endif
}

// ---------- conversion / packing kernels ----------

// Pack f32 W[K][N] into bf16 fragment layout dst[nt][kc][lane][16] where
// n = nt*16 + lane%16, k = kc*32 + 8*(lane/16) + (i%8) + 16*(i/8).
__global__ void wpack_kernel(const float* __restrict__ src, __bf16* __restrict__ dst,
                             int K, int N) {
  const int idx = blockIdx.x * blockDim.x + threadIdx.x;
  if (idx >= K * N) return;
  const int KC = K >> 5;
  const int i    = idx & 15;
  const int lane = (idx >> 4) & 31;
  const int rest = idx >> 9;
  const int kc   = rest % KC;
  const int nt   = rest / KC;
  const int n = nt * 16 + (lane & 15);
  const int k = kc * 32 + ((lane >> 4) << 3) + (i & 7) + ((i >> 3) << 4);
  dst[idx] = (__bf16)src[k * N + n];
}

// x [B][S][IN] f32 -> xb [S][B][IN] bf16 (time-major for the GRU)
__global__ void xconv_kernel(const float* __restrict__ x, __bf16* __restrict__ xb) {
  int idx = blockIdx.x * blockDim.x + threadIdx.x;
  if (idx >= B_ * S_ * IN_) return;
  int i  = idx & (IN_ - 1);
  int bs = idx >> 8;           // / IN_
  int s  = bs & (S_ - 1);
  int b  = bs >> 9;            // / S_
  xb[((size_t)s * B_ + b) * IN_ + i] = (__bf16)x[idx];
}

__global__ void zero_kernel(unsigned int* __restrict__ p, int n) {
  int i = blockIdx.x * blockDim.x + threadIdx.x;
  if (i < n) p[i] = 0u;
}

__global__ void pconv_kernel(const unsigned int* __restrict__ p,
                             __bf16* __restrict__ pb, int n) {
  int i = blockIdx.x * blockDim.x + threadIdx.x;
  if (i < n) pb[i] = (__bf16)__uint_as_float(p[i]);
}

// ---------- recurrent bidirectional GRU: one workgroup per direction ----------

__global__ __launch_bounds__(1024) void gru_kernel(
    const __bf16* __restrict__ xb,
    const __bf16* __restrict__ WgP0, const float* __restrict__ bg0,
    const __bf16* __restrict__ WcP0, const float* __restrict__ bc0,
    __bf16* __restrict__ out0,
    const __bf16* __restrict__ WgP1, const float* __restrict__ bg1,
    const __bf16* __restrict__ WcP1, const float* __restrict__ bc1,
    __bf16* __restrict__ out1) {
  extern __shared__ __bf16 smem[];
  __bf16* h_s  = smem;               // [32][HP_]
  __bf16* rh_s = smem + B_ * HP_;    // [32][HP_]

  const int dir = blockIdx.x;
  const __bf16* WgP = dir ? WgP1 : WgP0;
  const float*  bg  = dir ? bg1  : bg0;
  const __bf16* WcP = dir ? WcP1 : WcP0;
  const float*  bc  = dir ? bc1  : bc0;
  __bf16* out = dir ? out1 : out0;

  for (int i = threadIdx.x; i < B_ * HP_; i += 1024)
    h_s[i] = (__bf16)0.0f;
  __syncthreads();

  const int lane  = threadIdx.x & 31;
  const int w     = threadIdx.x >> 5;  // wave id 0..31 == H column tile
  const int arow  = lane & 15;
  const int ahalf = lane >> 4;
  const int col   = w * 16 + arow;     // this lane's output column in H

  const float bgr = bg[col];
  const float bgu = bg[H_ + col];
  const float bcc = bc[col];

#pragma unroll 1
  for (int t = 0; t < S_; ++t) {
    const int s = dir ? (S_ - 1 - t) : t;
    const __bf16* xs = xb + (size_t)s * B_ * IN_;

    // prefetch next timestep's x slice (8192 bf16 across 1024 threads)
    {
      const int sn = dir ? (s > 0 ? s - 1 : 0) : (s < S_ - 1 ? s + 1 : s);
      __builtin_prefetch(xb + (size_t)sn * B_ * IN_ + threadIdx.x * 8, 0, 1);
    }

    float uacc0[8], uacc1[8];

    // ---- pass 1: gates ru = sigmoid([x,h] @ Wg + bg) ----
    {
      v8f accr0 = {}, accr1 = {}, accu0 = {}, accu1 = {};
#pragma unroll 1
      for (int kc = 0; kc < 8; ++kc) {           // x part (global)
        v16bf a0 = load_a16(xs + arow * IN_ + kc * 32 + ahalf * 8);
        v16bf a1 = load_a16(xs + (16 + arow) * IN_ + kc * 32 + ahalf * 8);
        v16bf br = load_bP(WgP, 24, w, kc);
        v16bf bu = load_bP(WgP, 24, 32 + w, kc);
        accr0 = wmma_bf16(a0, br, accr0);
        accr1 = wmma_bf16(a1, br, accr1);
        accu0 = wmma_bf16(a0, bu, accu0);
        accu1 = wmma_bf16(a1, bu, accu1);
      }
#pragma unroll 2
      for (int kc = 8; kc < 24; ++kc) {          // h part (LDS)
        v16bf a0 = load_a16(&h_s[arow * HP_ + kc * 32 - IN_ + ahalf * 8]);
        v16bf a1 = load_a16(&h_s[(16 + arow) * HP_ + kc * 32 - IN_ + ahalf * 8]);
        v16bf br = load_bP(WgP, 24, w, kc);
        v16bf bu = load_bP(WgP, 24, 32 + w, kc);
        accr0 = wmma_bf16(a0, br, accr0);
        accr1 = wmma_bf16(a1, br, accr1);
        accu0 = wmma_bf16(a0, bu, accu0);
        accu1 = wmma_bf16(a1, bu, accu1);
      }
#pragma unroll
      for (int v = 0; v < 8; ++v) {
        const int row0 = v + 8 * ahalf;
        const int row1 = 16 + row0;
        const float r0 = sigm(accr0[v] + bgr);
        const float r1 = sigm(accr1[v] + bgr);
        uacc0[v] = sigm(accu0[v] + bgu);
        uacc1[v] = sigm(accu1[v] + bgu);
        rh_s[row0 * HP_ + col] = (__bf16)(r0 * (float)h_s[row0 * HP_ + col]);
        rh_s[row1 * HP_ + col] = (__bf16)(r1 * (float)h_s[row1 * HP_ + col]);
      }
    }
    __syncthreads();

    // ---- pass 2: candidate c = tanh([x, r*h] @ Wc + bc), state update ----
    {
      v8f accc0 = {}, accc1 = {};
#pragma unroll 1
      for (int kc = 0; kc < 8; ++kc) {           // x part (global)
        v16bf a0 = load_a16(xs + arow * IN_ + kc * 32 + ahalf * 8);
        v16bf a1 = load_a16(xs + (16 + arow) * IN_ + kc * 32 + ahalf * 8);
        v16bf bb = load_bP(WcP, 24, w, kc);
        accc0 = wmma_bf16(a0, bb, accc0);
        accc1 = wmma_bf16(a1, bb, accc1);
      }
#pragma unroll 2
      for (int kc = 8; kc < 24; ++kc) {          // r*h part (LDS)
        v16bf a0 = load_a16(&rh_s[arow * HP_ + kc * 32 - IN_ + ahalf * 8]);
        v16bf a1 = load_a16(&rh_s[(16 + arow) * HP_ + kc * 32 - IN_ + ahalf * 8]);
        v16bf bb = load_bP(WcP, 24, w, kc);
        accc0 = wmma_bf16(a0, bb, accc0);
        accc1 = wmma_bf16(a1, bb, accc1);
      }
#pragma unroll
      for (int v = 0; v < 8; ++v) {
        const int row0 = v + 8 * ahalf;
        const int row1 = 16 + row0;
        const float c0 = fast_tanh(accc0[v] + bcc);
        const float c1 = fast_tanh(accc1[v] + bcc);
        const float hn0 = uacc0[v] * (float)h_s[row0 * HP_ + col] + (1.0f - uacc0[v]) * c0;
        const float hn1 = uacc1[v] * (float)h_s[row1 * HP_ + col] + (1.0f - uacc1[v]) * c1;
        h_s[row0 * HP_ + col] = (__bf16)hn0;
        h_s[row1 * HP_ + col] = (__bf16)hn1;
        out[((size_t)s * B_ + row0) * H_ + col] = (__bf16)hn0;
        out[((size_t)s * B_ + row1) * H_ + col] = (__bf16)hn1;
      }
    }
    __syncthreads();
  }
}

// ---------- fused FC + relu + max-pool over S ----------
// Rows r = s*32 + b; A gathered from shifted out_fw / x / out_bw regions.
// Each wave: one 16-row M tile x four 16-col N tiles (4x A reuse).

__global__ __launch_bounds__(256) void fc_pool_kernel(
    const __bf16* __restrict__ xb,
    const __bf16* __restrict__ out_fw,
    const __bf16* __restrict__ out_bw,
    const __bf16* __restrict__ fcwP,   // packed [32][40][32][16]
    const float* __restrict__ fc_b,
    unsigned int* __restrict__ pooled) {
  const int lane  = threadIdx.x & 31;
  const int arow  = lane & 15;
  const int ahalf = lane >> 4;
  const int gw = blockIdx.x * 8 + (threadIdx.x >> 5);
  const int tm  = gw >> 3;        // 0..1023 (M tiles, M=16384)
  const int tnb = (gw & 7) * 4;   // N tile group base: 0,4,...,28
  const int row_base = tm * 16;
  const int r = row_base + arow;
  const int s = r >> 5;           // uniform within the wave
  const int b = r & 31;

  v8f acc0 = {}, acc1 = {}, acc2 = {}, acc3 = {};

#pragma unroll 1
  for (int kc = 0; kc < 16; ++kc) {              // c_left = out_fw[s-1]
    v16bf a = (s == 0) ? zero16()
              : load_a16(out_fw + ((size_t)(s - 1) * B_ + b) * H_ + kc * 32 + ahalf * 8);
    acc0 = wmma_bf16(a, load_bP(fcwP, 40, tnb + 0, kc), acc0);
    acc1 = wmma_bf16(a, load_bP(fcwP, 40, tnb + 1, kc), acc1);
    acc2 = wmma_bf16(a, load_bP(fcwP, 40, tnb + 2, kc), acc2);
    acc3 = wmma_bf16(a, load_bP(fcwP, 40, tnb + 3, kc), acc3);
  }
#pragma unroll 1
  for (int kc = 16; kc < 24; ++kc) {             // x[s]
    v16bf a = load_a16(xb + ((size_t)s * B_ + b) * IN_ + (kc - 16) * 32 + ahalf * 8);
    acc0 = wmma_bf16(a, load_bP(fcwP, 40, tnb + 0, kc), acc0);
    acc1 = wmma_bf16(a, load_bP(fcwP, 40, tnb + 1, kc), acc1);
    acc2 = wmma_bf16(a, load_bP(fcwP, 40, tnb + 2, kc), acc2);
    acc3 = wmma_bf16(a, load_bP(fcwP, 40, tnb + 3, kc), acc3);
  }
#pragma unroll 1
  for (int kc = 24; kc < 40; ++kc) {             // c_right = out_bw[s+1]
    v16bf a = (s == S_ - 1) ? zero16()
              : load_a16(out_bw + ((size_t)(s + 1) * B_ + b) * H_ + (kc - 24) * 32 + ahalf * 8);
    acc0 = wmma_bf16(a, load_bP(fcwP, 40, tnb + 0, kc), acc0);
    acc1 = wmma_bf16(a, load_bP(fcwP, 40, tnb + 1, kc), acc1);
    acc2 = wmma_bf16(a, load_bP(fcwP, 40, tnb + 2, kc), acc2);
    acc3 = wmma_bf16(a, load_bP(fcwP, 40, tnb + 3, kc), acc3);
  }

#pragma unroll
  for (int j = 0; j < 4; ++j) {
    const v8f acc = (j == 0) ? acc0 : (j == 1) ? acc1 : (j == 2) ? acc2 : acc3;
    const int col = (tnb + j) * 16 + arow;
    const float bias = fc_b[col];
#pragma unroll
    for (int v = 0; v < 8; ++v) {
      const int rr = row_base + v + 8 * ahalf;
      const int bb = rr & 31;
      float val = acc[v] + bias;
      val = val > 0.0f ? val : 0.0f;     // relu >= 0 -> uint-bit max is valid
      atomicMax(pooled + bb * H_ + col, __float_as_uint(val));
    }
  }
}

// ---------- MLP head: [32,512] @ [512,256] + bias ----------

__global__ __launch_bounds__(1024) void head_kernel(
    const __bf16* __restrict__ pooled_b,  // [32][512]
    const __bf16* __restrict__ mlpP,      // packed [16][16][32][16]
    const float* __restrict__ mlp_b,
    float* __restrict__ out) {            // [32][256]
  const int lane  = threadIdx.x & 31;
  const int w     = threadIdx.x >> 5;
  const int arow  = lane & 15;
  const int ahalf = lane >> 4;
  const int tm = w >> 4;   // 0..1
  const int tn = w & 15;   // 0..15

  v8f acc = {};
#pragma unroll 1
  for (int kc = 0; kc < 16; ++kc) {
    v16bf a  = load_a16(pooled_b + (tm * 16 + arow) * H_ + kc * 32 + ahalf * 8);
    v16bf bf = load_bP(mlpP, 16, tn, kc);
    acc = wmma_bf16(a, bf, acc);
  }
  const int col = tn * 16 + arow;
  const float bias = mlp_b[col];
#pragma unroll
  for (int v = 0; v < 8; ++v) {
    const int row = tm * 16 + v + 8 * ahalf;
    out[row * OUT_ + col] = acc[v] + bias;
  }
}

extern "C" void kernel_launch(void* const* d_in, const int* in_sizes, int n_in,
                              void* d_out, int out_size, void* d_ws, size_t ws_size,
                              hipStream_t stream) {
  (void)in_sizes; (void)n_in; (void)out_size; (void)ws_size;

  const float* x     = (const float*)d_in[0];
  const float* fw_Wg = (const float*)d_in[1];
  const float* fw_bg = (const float*)d_in[2];
  const float* fw_Wc = (const float*)d_in[3];
  const float* fw_bc = (const float*)d_in[4];
  const float* bw_Wg = (const float*)d_in[5];
  const float* bw_bg = (const float*)d_in[6];
  const float* bw_Wc = (const float*)d_in[7];
  const float* bw_bc = (const float*)d_in[8];
  const float* fc_w  = (const float*)d_in[9];
  const float* fc_b  = (const float*)d_in[10];
  const float* mlp_w = (const float*)d_in[11];
  const float* mlp_b = (const float*)d_in[12];

  char* ws = (char*)d_ws;
  size_t off = 0;
  auto take = [&](size_t bytes) -> char* {
    char* p = ws + off;
    off += (bytes + 255) & ~(size_t)255;
    return p;
  };
  __bf16* xb     = (__bf16*)take((size_t)S_ * B_ * IN_ * 2);
  __bf16* fwWgP  = (__bf16*)take((size_t)2 * H_ * (IN_ + H_) * 2);
  __bf16* fwWcP  = (__bf16*)take((size_t)H_ * (IN_ + H_) * 2);
  __bf16* bwWgP  = (__bf16*)take((size_t)2 * H_ * (IN_ + H_) * 2);
  __bf16* bwWcP  = (__bf16*)take((size_t)H_ * (IN_ + H_) * 2);
  __bf16* fcwP   = (__bf16*)take((size_t)H_ * CTX_ * 2);
  __bf16* mlpP   = (__bf16*)take((size_t)OUT_ * H_ * 2);
  __bf16* out_fw = (__bf16*)take((size_t)S_ * B_ * H_ * 2);
  __bf16* out_bw = (__bf16*)take((size_t)S_ * B_ * H_ * 2);
  unsigned int* pooled = (unsigned int*)take((size_t)B_ * H_ * 4);
  __bf16* pooled_b = (__bf16*)take((size_t)B_ * H_ * 2);

  xconv_kernel<<<(B_ * S_ * IN_) / 256, 256, 0, stream>>>(x, xb);
  wpack_kernel<<<((IN_ + H_) * 2 * H_ + 255) / 256, 256, 0, stream>>>(fw_Wg, fwWgP, IN_ + H_, 2 * H_);
  wpack_kernel<<<((IN_ + H_) * H_ + 255) / 256, 256, 0, stream>>>(fw_Wc, fwWcP, IN_ + H_, H_);
  wpack_kernel<<<((IN_ + H_) * 2 * H_ + 255) / 256, 256, 0, stream>>>(bw_Wg, bwWgP, IN_ + H_, 2 * H_);
  wpack_kernel<<<((IN_ + H_) * H_ + 255) / 256, 256, 0, stream>>>(bw_Wc, bwWcP, IN_ + H_, H_);
  wpack_kernel<<<(CTX_ * H_ + 255) / 256, 256, 0, stream>>>(fc_w, fcwP, CTX_, H_);
  wpack_kernel<<<(H_ * OUT_ + 255) / 256, 256, 0, stream>>>(mlp_w, mlpP, H_, OUT_);
  zero_kernel<<<(B_ * H_ + 255) / 256, 256, 0, stream>>>(pooled, B_ * H_);

  const size_t gru_lds = (size_t)2 * B_ * HP_ * sizeof(__bf16);  // 66560 B
  gru_kernel<<<2, 1024, gru_lds, stream>>>(xb,
                                           fwWgP, fw_bg, fwWcP, fw_bc, out_fw,
                                           bwWgP, bw_bg, bwWcP, bw_bc, out_bw);

  fc_pool_kernel<<<1024, 256, 0, stream>>>(xb, out_fw, out_bw, fcwP, fc_b, pooled);

  pconv_kernel<<<(B_ * H_ + 255) / 256, 256, 0, stream>>>(pooled, pooled_b, B_ * H_);

  head_kernel<<<1, 1024, 0, stream>>>(pooled_b, mlpP, mlp_b, (float*)d_out);
}
